// LSTM_AD_20203526160735
// MI455X (gfx1250) — matrix operand order repaired and checked by reading, not compiled, not run on hardware
//
#include <hip/hip_runtime.h>

// Problem constants (match reference)
#define B_SZ   128
#define T_SZ   512
#define D_SZ   32
#define H_SZ   512
#define O_SZ   320          // D*OUT = 32*10
#define BH     (B_SZ * H_SZ)
#define NWG    32
#define TPB    256

typedef __attribute__((ext_vector_type(16))) _Float16 v16h;
typedef __attribute__((ext_vector_type(8)))  _Float16 v8h;
typedef __attribute__((ext_vector_type(8)))  float    v8f;

// ---------------------------------------------------------------- helpers ---

__device__ __forceinline__ v8f wmma32(v16h a, v16h b, v8f c) {
    // (neg_a, A, neg_b, B, c_mod, C, reuse_a, reuse_b)
    return __builtin_amdgcn_wmma_f32_16x16x32_f16(false, a, false, b, (short)0, c,
                                                  false, false);
}

// A-matrix (16x32 f16) lane layout: lane = m + 16*half; per lane two 8-halves
// runs at k = 8*half and k = 16 + 8*half.
__device__ __forceinline__ v16h load_a_f16(const _Float16* __restrict__ rowp,
                                           int kc, int half) {
    const v8h lo = *(const v8h*)(rowp + kc + 8 * half);
    const v8h hi = *(const v8h*)(rowp + kc + 16 + 8 * half);
    v16h r;
#pragma unroll
    for (int i = 0; i < 8; ++i) { r[i] = lo[i]; r[8 + i] = hi[i]; }
    return r;
}

// A-matrix built from f32 source (layer-1 input x), converted in-register.
__device__ __forceinline__ v16h load_a_f32(const float* __restrict__ rowp, int half) {
    v16h r;
#pragma unroll
    for (int i = 0; i < 8; ++i) r[i]     = (_Float16)rowp[8 * half + i];
#pragma unroll
    for (int i = 0; i < 8; ++i) r[8 + i] = (_Float16)rowp[16 + 8 * half + i];
    return r;
}

// B-matrix (32x16 f16): lane = n + 16*half; per lane two 8-halves runs at
// k = 8*half and k = 16 + 8*half (same as A).
// Source row = W[n0 + (lane&15)] (row-major, K contiguous). 16B aligned here.
__device__ __forceinline__ v16h load_b_f16(const _Float16* __restrict__ rowp,
                                           int kc, int half) {
    const v8h lo = *(const v8h*)(rowp + kc + 8 * half);
    const v8h hi = *(const v8h*)(rowp + kc + 16 + 8 * half);
    v16h r;
#pragma unroll
    for (int i = 0; i < 8; ++i) { r[i] = lo[i]; r[8 + i] = hi[i]; }
    return r;
}

__device__ __forceinline__ float sigm(float v) {
    return 1.0f / (1.0f + __expf(-v));
}

// Grid-wide barrier: cnt at bar[0], gen at bar[16] (separate cachelines).
__device__ __forceinline__ void grid_barrier(unsigned* bar, unsigned nwg) {
    __syncthreads();
    if (threadIdx.x == 0) {
        __threadfence();
        unsigned* cnt = bar;
        unsigned* gen = bar + 16;
        unsigned g = atomicAdd(gen, 0u);
        unsigned prev = atomicAdd(cnt, 1u);
        if (prev == nwg - 1u) {
            atomicExch(cnt, 0u);   // reset BEFORE release
            atomicAdd(gen, 1u);    // release
        } else {
            while (atomicAdd(gen, 0u) == g) __builtin_amdgcn_s_sleep(1);
        }
        __threadfence();
    }
    __syncthreads();
}

// ---------------------------------------------------------------- phases ----
// Tile decode: rt = tile & 7 (batch rows), ct = tile >> 3 (gate/hidden cols).
// With waveId == tile, the 8 waves of a workgroup share ct => identical weight
// (B) addresses serviced by the WGP$, distinct activation (A) rows. Weights are
// the 4x-larger operand, so this is the operand worth sharing.

// Layer-1: gates = h1_old @ Whh1^T + x_t @ Wih1^T + b ; update c1, write h1_new (f16)
__device__ __forceinline__ void lstm_phase1(
    int t, int waveId, int nWaves, int lane,
    const float* __restrict__ x,
    const _Float16* __restrict__ h1r, _Float16* __restrict__ h1w,
    float* __restrict__ c1,
    const _Float16* __restrict__ wih1, const _Float16* __restrict__ whh1,
    const float* __restrict__ b_ih1, const float* __restrict__ b_hh1)
{
    const int half = lane >> 4, l16 = lane & 15;
    for (int tile = waveId; tile < (B_SZ / 16) * (H_SZ / 16); tile += nWaves) {
        const int m0 = (tile & 7) * 16;
        const int n0 = (tile >> 3) * 16;
        const int arow = m0 + l16;
        const int brow = n0 + l16;
        v8f acc[4] = {};
        const _Float16* aB = h1r + arow * H_SZ;
#pragma unroll 2
        for (int kc = 0; kc < H_SZ; kc += 32) {
            v16h a = load_a_f16(aB, kc, half);
#pragma unroll
            for (int g = 0; g < 4; ++g) {
                v16h b = load_b_f16(whh1 + (size_t)(g * H_SZ + brow) * H_SZ, kc, half);
                acc[g] = wmma32(a, b, acc[g]);
            }
        }
        {   // input contribution, K = D = 32 (one WMMA k-step per gate)
            v16h a = load_a_f32(x + ((size_t)arow * T_SZ + t) * D_SZ, half);
#pragma unroll
            for (int g = 0; g < 4; ++g) {
                v16h b = load_b_f16(wih1 + (size_t)(g * H_SZ + brow) * D_SZ, 0, half);
                acc[g] = wmma32(a, b, acc[g]);
            }
        }
        const int col = n0 + l16;
        const float bi = b_ih1[col]            + b_hh1[col];
        const float bf = b_ih1[H_SZ + col]     + b_hh1[H_SZ + col];
        const float bg = b_ih1[2 * H_SZ + col] + b_hh1[2 * H_SZ + col];
        const float bo = b_ih1[3 * H_SZ + col] + b_hh1[3 * H_SZ + col];
#pragma unroll
        for (int r = 0; r < 8; ++r) {
            const int row = m0 + r + 8 * half;
            const int idx = row * H_SZ + col;
            float gi = sigm(acc[0][r] + bi);
            float gf = sigm(acc[1][r] + bf);
            float gg = tanhf(acc[2][r] + bg);
            float go = sigm(acc[3][r] + bo);
            float cn = gf * c1[idx] + gi * gg;
            c1[idx] = cn;
            h1w[idx] = (_Float16)(go * tanhf(cn));
        }
    }
}

// Layer-2: gates = h1_new @ Wih2^T + h2_old @ Whh2^T + b ; update c2, write h2_new
__device__ __forceinline__ void lstm_phase2(
    int waveId, int nWaves, int lane,
    const _Float16* __restrict__ h1n,
    const _Float16* __restrict__ h2r, _Float16* __restrict__ h2w,
    float* __restrict__ c2,
    const _Float16* __restrict__ wih2, const _Float16* __restrict__ whh2,
    const float* __restrict__ b_ih2, const float* __restrict__ b_hh2)
{
    const int half = lane >> 4, l16 = lane & 15;
    for (int tile = waveId; tile < (B_SZ / 16) * (H_SZ / 16); tile += nWaves) {
        const int m0 = (tile & 7) * 16;
        const int n0 = (tile >> 3) * 16;
        const int arow = m0 + l16;
        const int brow = n0 + l16;
        v8f acc[4] = {};
        const _Float16* a1B = h1n + arow * H_SZ;
        const _Float16* a2B = h2r + arow * H_SZ;
#pragma unroll 2
        for (int kc = 0; kc < H_SZ; kc += 32) {
            v16h a1 = load_a_f16(a1B, kc, half);
            v16h a2 = load_a_f16(a2B, kc, half);
#pragma unroll
            for (int g = 0; g < 4; ++g) {
                v16h b1 = load_b_f16(wih2 + (size_t)(g * H_SZ + brow) * H_SZ, kc, half);
                acc[g] = wmma32(a1, b1, acc[g]);
                v16h b2 = load_b_f16(whh2 + (size_t)(g * H_SZ + brow) * H_SZ, kc, half);
                acc[g] = wmma32(a2, b2, acc[g]);
            }
        }
        const int col = n0 + l16;
        const float bi = b_ih2[col]            + b_hh2[col];
        const float bf = b_ih2[H_SZ + col]     + b_hh2[H_SZ + col];
        const float bg = b_ih2[2 * H_SZ + col] + b_hh2[2 * H_SZ + col];
        const float bo = b_ih2[3 * H_SZ + col] + b_hh2[3 * H_SZ + col];
#pragma unroll
        for (int r = 0; r < 8; ++r) {
            const int row = m0 + r + 8 * half;
            const int idx = row * H_SZ + col;
            float gi = sigm(acc[0][r] + bi);
            float gf = sigm(acc[1][r] + bf);
            float gg = tanhf(acc[2][r] + bg);
            float go = sigm(acc[3][r] + bo);
            float cn = gf * c2[idx] + gi * gg;
            c2[idx] = cn;
            h2w[idx] = (_Float16)(go * tanhf(cn));
        }
    }
}

// Head: out_t = h2_new @ Wlin^T + b_lin   (128 x 320, K = 512)
__device__ __forceinline__ void lstm_phase3(
    int t, int waveId, int nWaves, int lane,
    const _Float16* __restrict__ h2n,
    const _Float16* __restrict__ wlin, const float* __restrict__ b_lin,
    float* __restrict__ out)
{
    const int half = lane >> 4, l16 = lane & 15;
    for (int tile = waveId; tile < (B_SZ / 16) * (O_SZ / 16); tile += nWaves) {
        const int m0 = (tile & 7) * 16;
        const int n0 = (tile >> 3) * 16;
        const int arow = m0 + l16;
        const int brow = n0 + l16;
        v8f acc = {};
        const _Float16* aB = h2n + arow * H_SZ;
#pragma unroll 2
        for (int kc = 0; kc < H_SZ; kc += 32) {
            v16h a = load_a_f16(aB, kc, half);
            v16h b = load_b_f16(wlin + (size_t)brow * H_SZ, kc, half);
            acc = wmma32(a, b, acc);
        }
        const int col = n0 + l16;
        const float bl = b_lin[col];
#pragma unroll
        for (int r = 0; r < 8; ++r) {
            const int row = m0 + r + 8 * half;
            out[((size_t)row * T_SZ + t) * O_SZ + col] = acc[r] + bl;
        }
    }
}

// ---------------------------------------------------------------- kernels ---

__global__ void zero_bar_kernel(unsigned* bar) {
    if (threadIdx.x < 64) bar[threadIdx.x] = 0u;
}

__global__ void cvt_f32_f16_kernel(const float* __restrict__ s,
                                   _Float16* __restrict__ d, int n) {
    int i = blockIdx.x * blockDim.x + threadIdx.x;
    int st = gridDim.x * blockDim.x;
    for (; i < n; i += st) d[i] = (_Float16)s[i];
}

__global__ __launch_bounds__(TPB) void lstm_persistent_kernel(
    const float* __restrict__ x,
    const float* __restrict__ b_ih1, const float* __restrict__ b_hh1,
    const float* __restrict__ b_ih2, const float* __restrict__ b_hh2,
    const float* __restrict__ b_lin,
    const _Float16* __restrict__ wih1, const _Float16* __restrict__ whh1,
    const _Float16* __restrict__ wih2, const _Float16* __restrict__ whh2,
    const _Float16* __restrict__ wlin,
    _Float16* h1buf, _Float16* h2buf, float* c1, float* c2,
    unsigned* bar, float* out)
{
    const int lane   = threadIdx.x & 31;
    const int waveId = blockIdx.x * (blockDim.x >> 5) + (threadIdx.x >> 5);
    const int nWaves = gridDim.x * (blockDim.x >> 5);

    // Zero initial state (buffer 0 of h1/h2, plus c1/c2).
    {
        int tid  = blockIdx.x * blockDim.x + threadIdx.x;
        int nthr = gridDim.x * blockDim.x;
        for (int i = tid; i < BH; i += nthr) {
            h1buf[i] = (_Float16)0.0f;
            h2buf[i] = (_Float16)0.0f;
            c1[i] = 0.0f;
            c2[i] = 0.0f;
        }
    }
    grid_barrier(bar, gridDim.x);

    // Prologue: layer-1 step 0 (reads h1 buf0, writes h1 buf1).
    lstm_phase1(0, waveId, nWaves, lane, x, h1buf, h1buf + BH, c1,
                wih1, whh1, b_ih1, b_hh1);

#pragma unroll 1
    for (int t = 0; t < T_SZ; ++t) {
        const _Float16* h1n = h1buf + (((t + 1) & 1) ? BH : 0); // written by phase1(t)
        const _Float16* h2r = h2buf + ((t & 1) ? BH : 0);
        _Float16*       h2w = h2buf + (((t + 1) & 1) ? BH : 0);

        grid_barrier(bar, gridDim.x);               // h1(t) visible everywhere
        lstm_phase2(waveId, nWaves, lane, h1n, h2r, h2w, c2,
                    wih2, whh2, b_ih2, b_hh2);
        grid_barrier(bar, gridDim.x);               // h2(t) visible everywhere

        // Head for step t overlaps with layer-1 for step t+1 (independent).
        lstm_phase3(t, waveId, nWaves, lane, h2w, wlin, b_lin, out);
        if (t + 1 < T_SZ) {
            _Float16* h1w = h1buf + ((t & 1) ? BH : 0);
            lstm_phase1(t + 1, waveId, nWaves, lane, x, h1n, h1w, c1,
                        wih1, whh1, b_ih1, b_hh1);
        }
    }
}

// ---------------------------------------------------------------- launch ----

extern "C" void kernel_launch(void* const* d_in, const int* in_sizes, int n_in,
                              void* d_out, int out_size, void* d_ws, size_t ws_size,
                              hipStream_t stream) {
    const float* x     = (const float*)d_in[0];
    const float* W_ih1 = (const float*)d_in[1];
    const float* W_hh1 = (const float*)d_in[2];
    const float* b_ih1 = (const float*)d_in[3];
    const float* b_hh1 = (const float*)d_in[4];
    const float* W_ih2 = (const float*)d_in[5];
    const float* W_hh2 = (const float*)d_in[6];
    const float* b_ih2 = (const float*)d_in[7];
    const float* b_hh2 = (const float*)d_in[8];
    const float* W_lin = (const float*)d_in[9];
    const float* b_lin = (const float*)d_in[10];
    float* out = (float*)d_out;

    // Workspace layout (all offsets 256B-aligned).
    char* ws = (char*)d_ws;
    size_t off = 0;
    unsigned* bar = (unsigned*)(ws + off);      off += 256;
    _Float16* h1buf = (_Float16*)(ws + off);    off += (size_t)2 * BH * 2;
    _Float16* h2buf = (_Float16*)(ws + off);    off += (size_t)2 * BH * 2;
    float*    c1    = (float*)(ws + off);       off += (size_t)BH * 4;
    float*    c2    = (float*)(ws + off);       off += (size_t)BH * 4;
    _Float16* wih1h = (_Float16*)(ws + off);    off += (size_t)4 * H_SZ * D_SZ * 2;
    _Float16* whh1h = (_Float16*)(ws + off);    off += (size_t)4 * H_SZ * H_SZ * 2;
    _Float16* wih2h = (_Float16*)(ws + off);    off += (size_t)4 * H_SZ * H_SZ * 2;
    _Float16* whh2h = (_Float16*)(ws + off);    off += (size_t)4 * H_SZ * H_SZ * 2;
    _Float16* wlinh = (_Float16*)(ws + off);    off += (size_t)O_SZ * H_SZ * 2;

    zero_bar_kernel<<<1, 64, 0, stream>>>(bar);

    cvt_f32_f16_kernel<<<256, TPB, 0, stream>>>(W_ih1, wih1h, 4 * H_SZ * D_SZ);
    cvt_f32_f16_kernel<<<512, TPB, 0, stream>>>(W_hh1, whh1h, 4 * H_SZ * H_SZ);
    cvt_f32_f16_kernel<<<512, TPB, 0, stream>>>(W_ih2, wih2h, 4 * H_SZ * H_SZ);
    cvt_f32_f16_kernel<<<512, TPB, 0, stream>>>(W_hh2, whh2h, 4 * H_SZ * H_SZ);
    cvt_f32_f16_kernel<<<256, TPB, 0, stream>>>(W_lin, wlinh, O_SZ * H_SZ);

    lstm_persistent_kernel<<<NWG, TPB, 0, stream>>>(
        x, b_ih1, b_hh1, b_ih2, b_hh2, b_lin,
        wih1h, whh1h, wih2h, whh2h, wlinh,
        h1buf, h2buf, c1, c2, bar, out);

    (void)in_sizes; (void)n_in; (void)out_size; (void)ws_size;
}
